// HeteroGNN_31121333027532
// MI455X (gfx1250) — matrix operand order repaired and checked by reading, not compiled, not run on hardware
//
#include <hip/hip_runtime.h>
#include <hip/hip_bf16.h>

typedef __attribute__((ext_vector_type(16))) _Float16 v16h;
typedef __attribute__((ext_vector_type(8)))  float    v8f;
typedef __attribute__((ext_vector_type(4)))  float    v4f;

#define NC_N 100000
#define NP_N 50000
#define NS_N 5000
#define CCH 128

// ---------------- input projection: out[i,c] = sum_k x[i,k]*W[c,k] + b[c] ----------------
__global__ void encode_kernel(const float* __restrict__ x, const float* __restrict__ W,
                              const float* __restrict__ b, float* __restrict__ out,
                              int n, int K) {
  int i = blockIdx.x;
  int c = threadIdx.x;
  if (i >= n) return;
  float s = b[c];
  for (int k = 0; k < K; ++k) s = fmaf(x[(size_t)i * K + k], W[c * K + k], s);
  out[(size_t)i * CCH + c] = s;
}

// edge-type pairs per node type (0=customer,1=product,2=store)
__device__ __constant__ int PAIR_A[3] = {1, 0, 2};
__device__ __constant__ int PAIR_B[3] = {3, 5, 4};

// ---------------- weight prep: f16 copies of Wl, f16 summed Wr pairs, summed biases ------
__global__ void prep_weights_kernel(const float* __restrict__ Wl, const float* __restrict__ Wr,
                                    const float* __restrict__ bl,
                                    _Float16* __restrict__ Wlh, _Float16* __restrict__ Wrh,
                                    float* __restrict__ bsum) {
  int idx = blockIdx.x * blockDim.x + threadIdx.x;
  if (idx < 12 * 16384) Wlh[idx] = (_Float16)Wl[idx];
  if (idx < 6 * 16384) {
    int mat = idx >> 14, i = idx & 16383;
    int l = mat / 3, t = mat - 3 * l;
    Wrh[idx] = (_Float16)(Wr[(size_t)(l * 6 + PAIR_A[t]) * 16384 + i] +
                          Wr[(size_t)(l * 6 + PAIR_B[t]) * 16384 + i]);
  }
  if (idx < 6 * 128) {
    int vec = idx >> 7, i = idx & 127;
    int l = vec / 3, t = vec - 3 * l;
    bsum[idx] = bl[(l * 6 + PAIR_A[t]) * 128 + i] + bl[(l * 6 + PAIR_B[t]) * 128 + i];
  }
}

// ---------------- scatter-add: msg[dst,:] += x[src,:], cnt[dst] += 1 ---------------------
__global__ void scatter_kernel(const float* __restrict__ x, const int* __restrict__ src,
                               const int* __restrict__ dst, float* __restrict__ msg,
                               float* __restrict__ cnt, int E) {
  int e = blockIdx.x * 2 + (threadIdx.x >> 7);
  if (e >= E) return;
  int c = threadIdx.x & 127;
  int s = src[e], d = dst[e];
  atomicAdd(&msg[(size_t)d * CCH + c], x[(size_t)s * CCH + c]);
  if (c == 0) atomicAdd(&cnt[d], 1.0f);
}

// A-tile load (16x32 f16 layout): lane = row (l&15), lane group (l>>4) selects contiguous K-half (16 elems).
__device__ inline v16h load_a_f16(const float* __restrict__ X, int row, int k, int g,
                                  float scale) {
  const float* p = X + (size_t)row * CCH + k * 32 + g * 16;
  v4f a0 = *(const v4f*)(p);
  v4f a1 = *(const v4f*)(p + 4);
  v4f a2 = *(const v4f*)(p + 8);
  v4f a3 = *(const v4f*)(p + 12);
  v16h r;
#pragma unroll
  for (int i = 0; i < 4; ++i) {
    r[i]      = (_Float16)(a0[i] * scale);
    r[4 + i]  = (_Float16)(a1[i] * scale);
    r[8 + i]  = (_Float16)(a2[i] * scale);
    r[12 + i] = (_Float16)(a3[i] * scale);
  }
  return r;
}

// ---------------- fused SAGE update: 3 GEMMs (WMMA) + bias + LayerNorm + ReLU ------------
// out = relu(LN(msg1@Wl1^T + msg2@Wl2^T + x@(Wr1+Wr2)^T + (bl1+bl2)))  [0.5 dropped: LN-invariant]
__global__ __launch_bounds__(256) void node_update_kernel(
    const float* __restrict__ msg1, const float* __restrict__ cnt1,
    const float* __restrict__ msg2, const float* __restrict__ cnt2,
    const float* __restrict__ xroot,
    const _Float16* __restrict__ Wl1h, const _Float16* __restrict__ Wl2h,
    const _Float16* __restrict__ Wrh,
    const float* __restrict__ bsum, const float* __restrict__ gamma,
    const float* __restrict__ beta, float* __restrict__ out, int nrows) {
  int lane = threadIdx.x & 31;
  int wave = threadIdx.x >> 5;
  int tile = blockIdx.x * 8 + wave;
  int ntiles = (nrows + 15) >> 4;
  if (tile >= ntiles) return;  // wave-uniform exit: live waves keep EXEC all-ones for WMMA
  int base = tile * 16;
  int nloc = lane & 15;
  int g = lane >> 4;
  int rowA = base + nloc;
  if (rowA >= nrows) rowA = base;  // tail clamp; clamped rows never stored

  v8f acc[8];
#pragma unroll
  for (int t = 0; t < 8; ++t) {
    float bv = bsum[t * 16 + nloc];
#pragma unroll
    for (int v = 0; v < 8; ++v) acc[t][v] = bv;
  }

  const float*    Xs[3] = {msg1, msg2, xroot};
  const float*    Cs[3] = {cnt1, cnt2, nullptr};
  const _Float16* Ws[3] = {Wl1h, Wl2h, Wrh};
#pragma unroll
  for (int m = 0; m < 3; ++m) {
    const float* X = Xs[m];
    const _Float16* W = Ws[m];
    float scale = 1.0f;
    if (Cs[m]) scale = 1.0f / fmaxf(Cs[m][rowA], 1.0f);  // fold mean divide into A
#pragma unroll
    for (int k = 0; k < 4; ++k) {
      v16h a = load_a_f16(X, rowA, k, g, scale);
#pragma unroll
      for (int t = 0; t < 8; ++t) {
        // B = W^T tile: lane nloc = output channel, contiguous 16 f16 along K
        v16h b = *(const v16h*)(W + (size_t)(t * 16 + nloc) * CCH + k * 32 + g * 16);
        acc[t] = __builtin_amdgcn_wmma_f32_16x16x32_f16(
            false, a, false, b, (short)0, acc[t], false, false);
      }
    }
  }

  // In-register LayerNorm over 128 channels per row: each row lives in one 16-lane half.
  v8f s, q;
#pragma unroll
  for (int v = 0; v < 8; ++v) { s[v] = 0.f; q[v] = 0.f; }
#pragma unroll
  for (int t = 0; t < 8; ++t)
#pragma unroll
    for (int v = 0; v < 8; ++v) { float a = acc[t][v]; s[v] += a; q[v] += a * a; }
#pragma unroll
  for (int v = 0; v < 8; ++v) {
    float sv = s[v], qv = q[v];
    sv += __shfl_xor(sv, 1); qv += __shfl_xor(qv, 1);
    sv += __shfl_xor(sv, 2); qv += __shfl_xor(qv, 2);
    sv += __shfl_xor(sv, 4); qv += __shfl_xor(qv, 4);
    sv += __shfl_xor(sv, 8); qv += __shfl_xor(qv, 8);
    float mu  = sv * (1.0f / 128.0f);
    float var = qv * (1.0f / 128.0f) - mu * mu;
    s[v] = mu;
    q[v] = rsqrtf(var + 1e-5f);
  }

#pragma unroll
  for (int t = 0; t < 8; ++t) {
    float gam = gamma[t * 16 + nloc];
    float bet = beta[t * 16 + nloc];
#pragma unroll
    for (int v = 0; v < 8; ++v) {
      int row = base + v + 8 * g;  // C/D layout: VGPR v -> M=v (+8 for upper half)
      if (row < nrows) {
        float val = (acc[t][v] - s[v]) * q[v] * gam + bet;
        out[(size_t)row * CCH + t * 16 + nloc] = fmaxf(val, 0.0f);
      }
    }
  }
}

extern "C" void kernel_launch(void* const* d_in, const int* in_sizes, int n_in,
                              void* d_out, int out_size, void* d_ws, size_t ws_size,
                              hipStream_t stream) {
  (void)n_in; (void)out_size; (void)ws_size;
  const float* x_customer  = (const float*)d_in[0];
  const float* x_product   = (const float*)d_in[1];
  const float* x_store     = (const float*)d_in[2];
  const int* ei_buys       = (const int*)d_in[3];
  const int* ei_bought_by  = (const int*)d_in[4];
  const int* ei_visits     = (const int*)d_in[5];
  const int* ei_visited_by = (const int*)d_in[6];
  const int* ei_sold_at    = (const int*)d_in[7];
  const int* ei_sells      = (const int*)d_in[8];
  const float* W_cust  = (const float*)d_in[9];
  const float* b_cust  = (const float*)d_in[10];
  const float* W_prod  = (const float*)d_in[11];
  const float* b_prod  = (const float*)d_in[12];
  const float* W_store = (const float*)d_in[13];
  const float* b_store = (const float*)d_in[14];
  const float* Wl    = (const float*)d_in[15];
  const float* bl    = (const float*)d_in[16];
  const float* Wr    = (const float*)d_in[17];
  const float* gamma = (const float*)d_in[18];
  const float* beta  = (const float*)d_in[19];

  const int E_buys = in_sizes[3] / 2, E_bb = in_sizes[4] / 2, E_vis = in_sizes[5] / 2,
            E_vb = in_sizes[6] / 2, E_sa = in_sizes[7] / 2, E_sl = in_sizes[8] / 2;

  char* w = (char*)d_ws;
  auto alloc = [&](size_t bytes) -> void* {
    void* p = (void*)w;
    w += (bytes + 255) & ~(size_t)255;
    return p;
  };
  float* xc = (float*)alloc((size_t)NC_N * CCH * 4);
  float* xp = (float*)alloc((size_t)NP_N * CCH * 4);
  float* xs = (float*)alloc((size_t)NS_N * CCH * 4);
  char* zstart = w;  // everything from here to zend is re-zeroed per layer
  float* msg_c1 = (float*)alloc((size_t)NC_N * CCH * 4);
  float* msg_c2 = (float*)alloc((size_t)NC_N * CCH * 4);
  float* msg_p1 = (float*)alloc((size_t)NP_N * CCH * 4);
  float* msg_p2 = (float*)alloc((size_t)NP_N * CCH * 4);
  float* msg_s1 = (float*)alloc((size_t)NS_N * CCH * 4);
  float* msg_s2 = (float*)alloc((size_t)NS_N * CCH * 4);
  float* cnt_c1 = (float*)alloc((size_t)NC_N * 4);
  float* cnt_c2 = (float*)alloc((size_t)NC_N * 4);
  float* cnt_p1 = (float*)alloc((size_t)NP_N * 4);
  float* cnt_p2 = (float*)alloc((size_t)NP_N * 4);
  float* cnt_s1 = (float*)alloc((size_t)NS_N * 4);
  float* cnt_s2 = (float*)alloc((size_t)NS_N * 4);
  size_t zbytes = (size_t)(w - zstart);
  _Float16* Wlh = (_Float16*)alloc((size_t)12 * 16384 * 2);
  _Float16* Wrh = (_Float16*)alloc((size_t)6 * 16384 * 2);
  float* bsum = (float*)alloc((size_t)6 * 128 * 4);

  encode_kernel<<<NC_N, 128, 0, stream>>>(x_customer, W_cust, b_cust, xc, NC_N, 6);
  encode_kernel<<<NP_N, 128, 0, stream>>>(x_product, W_prod, b_prod, xp, NP_N, 4);
  encode_kernel<<<NS_N, 128, 0, stream>>>(x_store, W_store, b_store, xs, NS_N, 3);
  prep_weights_kernel<<<(12 * 16384 + 255) / 256, 256, 0, stream>>>(Wl, Wr, bl, Wlh, Wrh, bsum);

  float* outc = (float*)d_out;
  float* outp = outc + (size_t)NC_N * CCH;
  float* outs = outp + (size_t)NP_N * CCH;

  for (int l = 0; l < 2; ++l) {
    hipMemsetAsync(zstart, 0, zbytes, stream);
    // all 6 scatters read old features of all types (must precede all updates)
    scatter_kernel<<<(E_buys + 1) / 2, 256, 0, stream>>>(xc, ei_buys, ei_buys + E_buys, msg_p1, cnt_p1, E_buys);
    scatter_kernel<<<(E_bb + 1) / 2, 256, 0, stream>>>(xp, ei_bought_by, ei_bought_by + E_bb, msg_c1, cnt_c1, E_bb);
    scatter_kernel<<<(E_vis + 1) / 2, 256, 0, stream>>>(xc, ei_visits, ei_visits + E_vis, msg_s1, cnt_s1, E_vis);
    scatter_kernel<<<(E_vb + 1) / 2, 256, 0, stream>>>(xs, ei_visited_by, ei_visited_by + E_vb, msg_c2, cnt_c2, E_vb);
    scatter_kernel<<<(E_sa + 1) / 2, 256, 0, stream>>>(xp, ei_sold_at, ei_sold_at + E_sa, msg_s2, cnt_s2, E_sa);
    scatter_kernel<<<(E_sl + 1) / 2, 256, 0, stream>>>(xs, ei_sells, ei_sells + E_sl, msg_p2, cnt_p2, E_sl);

    float* oc = (l == 1) ? outc : xc;
    float* op = (l == 1) ? outp : xp;
    float* os = (l == 1) ? outs : xs;
    {  // customer: edge types 1 (bought_by) & 3 (visited_by)
      int grid = (((NC_N + 15) / 16) + 7) / 8;
      node_update_kernel<<<grid, 256, 0, stream>>>(
          msg_c1, cnt_c1, msg_c2, cnt_c2, xc,
          Wlh + (size_t)(l * 6 + 1) * 16384, Wlh + (size_t)(l * 6 + 3) * 16384,
          Wrh + (size_t)(l * 3 + 0) * 16384,
          bsum + (l * 3 + 0) * 128, gamma + (l * 3 + 0) * 128, beta + (l * 3 + 0) * 128,
          oc, NC_N);
    }
    {  // product: edge types 0 (buys) & 5 (sells)
      int grid = (((NP_N + 15) / 16) + 7) / 8;
      node_update_kernel<<<grid, 256, 0, stream>>>(
          msg_p1, cnt_p1, msg_p2, cnt_p2, xp,
          Wlh + (size_t)(l * 6 + 0) * 16384, Wlh + (size_t)(l * 6 + 5) * 16384,
          Wrh + (size_t)(l * 3 + 1) * 16384,
          bsum + (l * 3 + 1) * 128, gamma + (l * 3 + 1) * 128, beta + (l * 3 + 1) * 128,
          op, NP_N);
    }
    {  // store: edge types 2 (visits) & 4 (sold_at)
      int grid = (((NS_N + 15) / 16) + 7) / 8;
      node_update_kernel<<<grid, 256, 0, stream>>>(
          msg_s1, cnt_s1, msg_s2, cnt_s2, xs,
          Wlh + (size_t)(l * 6 + 2) * 16384, Wlh + (size_t)(l * 6 + 4) * 16384,
          Wrh + (size_t)(l * 3 + 2) * 16384,
          bsum + (l * 3 + 2) * 128, gamma + (l * 3 + 2) * 128, beta + (l * 3 + 2) * 128,
          os, NS_N);
    }
  }
}
